// Decoder_74380243632630
// MI455X (gfx1250) — matrix operand order repaired and checked by reading, not compiled, not run on hardware
//
#include <hip/hip_runtime.h>

// ---------------------------------------------------------------------------
// Show-Attend-Tell decoder for MI455X (gfx1250, wave32, WMMA).
// Softmax over singleton axis => attention context z = sum_l enc_output
// (constant), so the attention stack is dead code.  GEMMs run on
// v_wmma_f32_16x16x32_bf16 with fp32 accumulation; weights packed once into
// WMMA fragment layout.  Vocab GEMM: one wave per N-tile computes BOTH
// M-tiles (B fragment loaded once -> 2 WMMAs), with the shared A operand
// staged in LDS via async load-to-LDS (ASYNCcnt) and read via ds_load_b128.
// ---------------------------------------------------------------------------

typedef __bf16 bf16;
typedef __attribute__((ext_vector_type(16))) __bf16 bf16x16;
typedef __attribute__((ext_vector_type(8)))  float  f32x8;

#define B_  32
#define L_  196
#define D_  512
#define T_  24
#define E_  512
#define N_  1024
#define V_  32000
#define G4_ 4096   // 4*N

__device__ __forceinline__ float sigmoidf_(float x) {
    return 1.0f / (1.0f + __expf(-x));
}

// ---- WMMA A-fragment (16x32 bf16) packed store ----------------------------
// Layout: lane = m + 16*(kl>>4);  q = kl & 15
__device__ __forceinline__ void storeA(bf16* dst, int KT, int mtile, int ktile,
                                       int m, int kl, float v) {
    int hi   = (kl >> 4) & 1;
    int q    = kl & 15;
    int lane = m + (hi << 4);
    dst[((size_t)((mtile * KT + ktile) << 5) + lane) * 16 + q] = (bf16)v;
}

// ---- Pack a row-major fp32 weight [nrows, K] into WMMA B fragments --------
// B[k,n] = W[n,k].  Layout: lane = (n&15) + 16*(kl>>4);  q = kl & 15
__global__ void pack_b_kernel(const float* __restrict__ src,
                              bf16* __restrict__ dst, int nrows, int K) {
    int e = blockIdx.x * blockDim.x + threadIdx.x;
    if (e >= nrows * K) return;
    int n = e / K, k = e - n * K;
    int ntile = n >> 4, nl = n & 15;
    int ktile = k >> 5, kl = k & 31;
    int lane  = nl + ((kl & 16) ? 16 : 0);
    int q     = kl & 15;
    int KT    = K >> 5;
    dst[((size_t)((ntile * KT + ktile) << 5) + lane) * 16 + q] = (bf16)src[e];
}

// ---- z[b,d] = sum_l enc[b,l,d]  (softmax over singleton axis == all ones) -
__global__ void z_kernel(const float* __restrict__ enc, float* __restrict__ z) {
    int idx = blockIdx.x * blockDim.x + threadIdx.x;
    if (idx >= B_ * D_) return;
    int b = idx >> 9, d = idx & 511;
    const float* p = enc + (size_t)b * L_ * D_ + d;
    float s = 0.f;
    for (int l = 0; l < L_; ++l) s += p[(size_t)l * D_];
    z[idx] = s;
}

// ---- h0 (packed bf16 A-frag) and c0 (fp32) from mean = z/L ----------------
__global__ void init_kernel(const float* __restrict__ z,
                            const float* __restrict__ hW, const float* __restrict__ hb,
                            const float* __restrict__ cW, const float* __restrict__ cb,
                            float* __restrict__ c, bf16* __restrict__ hP) {
    int idx = blockIdx.x * blockDim.x + threadIdx.x;
    if (idx >= B_ * N_) return;
    int b = idx >> 10, j = idx & 1023;
    const float* zb = z + (size_t)b * D_;
    const float* wh = hW + (size_t)j * D_;
    const float* wc = cW + (size_t)j * D_;
    const float inv = 1.0f / (float)L_;
    float sh = 0.f, sc = 0.f;
    for (int d = 0; d < D_; ++d) {
        float m = zb[d] * inv;
        sh += m * wh[d];
        sc += m * wc[d];
    }
    sh += hb[j];
    sc += cb[j];
    c[idx] = sc;
    storeA(hP, N_ >> 5, b >> 4, j >> 5, b & 15, j & 31, sh);
}

// ---- x_t = [emb[y_t], z]  -> packed bf16 A fragments (K = 1024) -----------
__global__ void packx_kernel(const float* __restrict__ emb,
                             const int* __restrict__ y,
                             const float* __restrict__ z,
                             bf16* __restrict__ xP, int t) {
    int idx = blockIdx.x * blockDim.x + threadIdx.x;
    if (idx >= B_ * (E_ + D_)) return;
    int b = idx >> 10, k = idx & 1023;
    float v;
    if (k < E_) {
        int tok = y[b * T_ + t];
        v = emb[(size_t)tok * E_ + k];
    } else {
        v = z[(size_t)b * D_ + (k - E_)];
    }
    storeA(xP, (E_ + D_) >> 5, b >> 4, k >> 5, b & 15, k & 31, v);
}

// ---- gates = x @ W_ih^T + h @ W_hh^T  (raw; biases added in cell) ---------
// One wave per 16-wide N-tile, BOTH M-tiles: each B fragment feeds 2 WMMAs.
__global__ void gates_kernel(const bf16* __restrict__ xP, const bf16* __restrict__ hP,
                             const bf16* __restrict__ wihP, const bf16* __restrict__ whhP,
                             float* __restrict__ gates) {
    int wave = blockIdx.x * (blockDim.x >> 5) + (threadIdx.x >> 5);
    int lane = threadIdx.x & 31;
    if (wave >= G4_ / 16) return;          // 256 N-tiles
    int ntile = wave;
    f32x8 acc0 = {}, acc1 = {};
    {
        const bf16* bp = wihP + (size_t)(ntile * 32) * 512 + lane * 16;
        const bf16* a0 = xP + (size_t)lane * 16;
        const bf16* a1 = xP + (size_t)32 * 512 + lane * 16;
#pragma unroll 4
        for (int kt = 0; kt < 32; ++kt) {
            bf16x16 bbv = *(const bf16x16*)(bp + (size_t)kt * 512);
            bf16x16 av0 = *(const bf16x16*)(a0 + (size_t)kt * 512);
            bf16x16 av1 = *(const bf16x16*)(a1 + (size_t)kt * 512);
            acc0 = __builtin_amdgcn_wmma_f32_16x16x32_bf16(
                false, av0, false, bbv, (short)0, acc0, false, false);
            acc1 = __builtin_amdgcn_wmma_f32_16x16x32_bf16(
                false, av1, false, bbv, (short)0, acc1, false, false);
        }
    }
    {
        const bf16* bp = whhP + (size_t)(ntile * 32) * 512 + lane * 16;
        const bf16* a0 = hP + (size_t)lane * 16;
        const bf16* a1 = hP + (size_t)32 * 512 + lane * 16;
#pragma unroll 4
        for (int kt = 0; kt < 32; ++kt) {
            bf16x16 bbv = *(const bf16x16*)(bp + (size_t)kt * 512);
            bf16x16 av0 = *(const bf16x16*)(a0 + (size_t)kt * 512);
            bf16x16 av1 = *(const bf16x16*)(a1 + (size_t)kt * 512);
            acc0 = __builtin_amdgcn_wmma_f32_16x16x32_bf16(
                false, av0, false, bbv, (short)0, acc0, false, false);
            acc1 = __builtin_amdgcn_wmma_f32_16x16x32_bf16(
                false, av1, false, bbv, (short)0, acc1, false, false);
        }
    }
    // C/D layout: VGPR r -> M = r + 8*(lane>=16), N = lane & 15
    int nl = lane & 15, hi = lane >> 4;
    int j  = ntile * 16 + nl;
#pragma unroll
    for (int r = 0; r < 8; ++r) {
        int m = hi * 8 + r;
        gates[(size_t)m * G4_ + j]        = acc0[r];
        gates[(size_t)(16 + m) * G4_ + j] = acc1[r];
    }
}

// ---- LSTM cell (torch gate order i,f,g,o); fuses bf16 A-packing of h_new --
__global__ void cell_kernel(const float* __restrict__ gates,
                            const float* __restrict__ bih, const float* __restrict__ bhh,
                            float* __restrict__ c, bf16* __restrict__ hP) {
    int idx = blockIdx.x * blockDim.x + threadIdx.x;
    if (idx >= B_ * N_) return;
    int b = idx >> 10, j = idx & 1023;
    const float* g = gates + (size_t)b * G4_;
    float ig = g[j]          + bih[j]          + bhh[j];
    float fg = g[N_ + j]     + bih[N_ + j]     + bhh[N_ + j];
    float gg = g[2 * N_ + j] + bih[2 * N_ + j] + bhh[2 * N_ + j];
    float og = g[3 * N_ + j] + bih[3 * N_ + j] + bhh[3 * N_ + j];
    float cn = sigmoidf_(fg) * c[idx] + sigmoidf_(ig) * tanhf(gg);
    float hn = sigmoidf_(og) * tanhf(cn);
    c[idx] = cn;
    storeA(hP, N_ >> 5, b >> 4, j >> 5, b & 15, j & 31, hn);
}

// ---- preds = h @ vocab_W^T + vocab_b -> out[b, v, t] ----------------------
// One wave per N-tile (2000 waves), BOTH M-tiles per wave (B read once from
// L2 -> 2 WMMAs).  Shared A operand (64 KB packed h fragments) is staged per
// block into LDS with global_load_async_to_lds_b128 (ASYNCcnt) and consumed
// through ds_load_b128.
__global__ void vocab_kernel(const bf16* __restrict__ hP, const bf16* __restrict__ vP,
                             const float* __restrict__ vb, float* __restrict__ out,
                             int t) {
    __shared__ bf16 sA[2 * 32 * 512];      // 64 KB: [mtile][ktile][lane][16]

    // Async-stage the full packed-h operand into LDS: 64 KB / (128 thr * 16 B)
    // = 32 b128 async copies per thread.
    {
        unsigned long long g = (unsigned long long)(uintptr_t)hP +
                               (unsigned long long)threadIdx.x * 16ull;
        unsigned int l = (unsigned int)(uintptr_t)(&sA[0]) +
                         (unsigned int)threadIdx.x * 16u;
#pragma unroll
        for (int i = 0; i < 32; ++i) {
            asm volatile("global_load_async_to_lds_b128 %0, %1, off"
                         :: "v"(l), "v"(g) : "memory");
            g += 128ull * 16ull;
            l += 128u * 16u;
        }
        asm volatile("s_wait_asynccnt 0x0" ::: "memory");
    }
    __syncthreads();

    int wave  = blockIdx.x * (blockDim.x >> 5) + (threadIdx.x >> 5);
    int lane  = threadIdx.x & 31;
    if (wave >= V_ / 16) return;           // 2000 N-tiles
    int ntile = wave;

    f32x8 acc0 = {}, acc1 = {};
    const bf16* bp = vP + (size_t)(ntile * 32) * 512 + lane * 16;
#pragma unroll 4
    for (int kt = 0; kt < 32; ++kt) {
        __builtin_prefetch(bp + (size_t)(kt + 1) * 512, 0, 1);  // global_prefetch_b8
        bf16x16 bbv = *(const bf16x16*)(bp + (size_t)kt * 512);
        bf16x16 av0 = *reinterpret_cast<const bf16x16*>(&sA[(size_t)kt * 512 + lane * 16]);
        bf16x16 av1 = *reinterpret_cast<const bf16x16*>(&sA[(size_t)(32 + kt) * 512 + lane * 16]);
        acc0 = __builtin_amdgcn_wmma_f32_16x16x32_bf16(
            false, av0, false, bbv, (short)0, acc0, false, false);
        acc1 = __builtin_amdgcn_wmma_f32_16x16x32_bf16(
            false, av1, false, bbv, (short)0, acc1, false, false);
    }
    int nl = lane & 15, hi = lane >> 4;
    int v  = ntile * 16 + nl;
    float bias = vb[v];
#pragma unroll
    for (int r = 0; r < 8; ++r) {
        int m = hi * 8 + r;
        out[((size_t)m * V_ + v) * T_ + t]        = acc0[r] + bias;
        out[((size_t)(16 + m) * V_ + v) * T_ + t] = acc1[r] + bias;
    }
}

// ---------------------------------------------------------------------------
extern "C" void kernel_launch(void* const* d_in, const int* in_sizes, int n_in,
                              void* d_out, int out_size, void* d_ws, size_t ws_size,
                              hipStream_t stream) {
    const float* enc      = (const float*)d_in[0];   // [B,L,D]
    const int*   y        = (const int*)  d_in[1];   // [B,T]
    const float* emb      = (const float*)d_in[2];   // [V,E]
    const float* W_ih     = (const float*)d_in[3];   // [4N, E+D]
    const float* W_hh     = (const float*)d_in[4];   // [4N, N]
    const float* b_ih     = (const float*)d_in[5];
    const float* b_hh     = (const float*)d_in[6];
    const float* init_h_W = (const float*)d_in[7];
    const float* init_h_b = (const float*)d_in[8];
    const float* init_c_W = (const float*)d_in[9];
    const float* init_c_b = (const float*)d_in[10];
    // d_in[11..14]: attention weights -- dead code (softmax over singleton axis)
    const float* vocab_W  = (const float*)d_in[15];  // [V, N]
    const float* vocab_b  = (const float*)d_in[16];
    float*       out      = (float*)d_out;           // [B, V, T]

    char* ws = (char*)d_ws;
    size_t off = 0;
    bf16*  vP   = (bf16*)(ws + off);  off += (size_t)2000 * 32 * 512 * sizeof(bf16); // 65.5 MB
    bf16*  wihP = (bf16*)(ws + off);  off += (size_t)256  * 32 * 512 * sizeof(bf16); // 8.4 MB
    bf16*  whhP = (bf16*)(ws + off);  off += (size_t)256  * 32 * 512 * sizeof(bf16); // 8.4 MB
    bf16*  xP   = (bf16*)(ws + off);  off += (size_t)2 * 32 * 512 * sizeof(bf16);
    bf16*  hP   = (bf16*)(ws + off);  off += (size_t)2 * 32 * 512 * sizeof(bf16);
    float* z    = (float*)(ws + off); off += (size_t)B_ * D_ * sizeof(float);
    float* c    = (float*)(ws + off); off += (size_t)B_ * N_ * sizeof(float);
    float* gts  = (float*)(ws + off); off += (size_t)B_ * G4_ * sizeof(float);
    (void)ws_size; (void)in_sizes; (void)n_in; (void)out_size;

    // One-time packs (deterministic; re-run every call)
    {
        int e = V_ * N_;
        pack_b_kernel<<<(e + 255) / 256, 256, 0, stream>>>(vocab_W, vP, V_, N_);
    }
    {
        int e = G4_ * (E_ + D_);
        pack_b_kernel<<<(e + 255) / 256, 256, 0, stream>>>(W_ih, wihP, G4_, E_ + D_);
    }
    {
        int e = G4_ * N_;
        pack_b_kernel<<<(e + 255) / 256, 256, 0, stream>>>(W_hh, whhP, G4_, N_);
    }

    z_kernel<<<(B_ * D_ + 255) / 256, 256, 0, stream>>>(enc, z);
    init_kernel<<<(B_ * N_ + 255) / 256, 256, 0, stream>>>(
        z, init_h_W, init_h_b, init_c_W, init_c_b, c, hP);

    for (int t = 0; t < T_; ++t) {
        packx_kernel<<<(B_ * (E_ + D_) + 255) / 256, 256, 0, stream>>>(emb, y, z, xP, t);
        // 256 N-tile waves, 4 waves (128 threads) per block
        gates_kernel<<<64, 128, 0, stream>>>(xP, hP, wihP, whhP, gts);
        cell_kernel<<<(B_ * N_ + 255) / 256, 256, 0, stream>>>(gts, b_ih, b_hh, c, hP);
        // 2000 N-tile waves, 4 per block
        vocab_kernel<<<500, 128, 0, stream>>>(hP, vP, vocab_b, out, t);
    }
}
